// DependencyParseModel_41609643163971
// MI455X (gfx1250) — matrix operand order repaired and checked by reading, not compiled, not run on hardware
//
#include <hip/hip_runtime.h>
#include <hip/hip_bf16.h>
#include <math.h>

#define SEQ 512
#define H   128
#define G4  512   // 4*H (gate width)
#define DD  256   // 2*H (bi-dir concat)
#define HID 512

typedef __attribute__((ext_vector_type(16))) __bf16        v16bf;
typedef __attribute__((ext_vector_type(8)))  float         v8f;
typedef __attribute__((ext_vector_type(8)))  unsigned short v8us;

union ABfrag { v16bf bf; v8us h[2]; };

__device__ __forceinline__ unsigned short f2bf(float f) {
  unsigned int u = __float_as_uint(f);
  u += 0x7FFFu + ((u >> 16) & 1u);   // round-to-nearest-even
  return (unsigned short)(u >> 16);
}

__device__ __forceinline__ float sigf(float x) { return 1.0f / (1.0f + __expf(-x)); }

// ---------------------------------------------------------------------------
// Embedding gather + concat, emit bf16 x[SEQ][H]
// ---------------------------------------------------------------------------
__global__ void embed_kernel(const int* __restrict__ wid, const int* __restrict__ tg,
                             const float* __restrict__ wtab, const float* __restrict__ ttab,
                             unsigned short* __restrict__ xbf) {
  int t = blockIdx.x, j = threadIdx.x;
  float v = (j < 100) ? wtab[(size_t)wid[t] * 100 + j]
                      : ttab[(size_t)tg[t]  * 28  + (j - 100)];
  xbf[(size_t)t * H + j] = f2bf(v);
}

// ---------------------------------------------------------------------------
// Generic f32 -> bf16 conversion
// ---------------------------------------------------------------------------
__global__ void cvt_f32_bf16(const float* __restrict__ src, unsigned short* __restrict__ dst, int n) {
  int i = blockIdx.x * blockDim.x + threadIdx.x;
  if (i < n) dst[i] = f2bf(src[i]);
}

// ---------------------------------------------------------------------------
// C[M,N] = Xbf[M,K] @ Wbf[N,K]^T (+bias0[n]+bias1[n]); one 16x16 tile / wave.
// Fragment layouts per CDNA5 ISA 7.12.2 (wave32).
// ---------------------------------------------------------------------------
__global__ void gemm_bf16_wmma(const unsigned short* __restrict__ X, int ldx,
                               const unsigned short* __restrict__ W, int ldw, int kofs,
                               const float* __restrict__ bias0, const float* __restrict__ bias1,
                               float* __restrict__ C, int ldc, int K, int tilesN) {
  int wave = threadIdx.x >> 5;
  int lane = threadIdx.x & 31;
  int tid  = blockIdx.x * (blockDim.x >> 5) + wave;
  int tm = (tid / tilesN) * 16;
  int tn = (tid % tilesN) * 16;
  int r  = lane & 15;
  int hi = lane >> 4;

  const unsigned short* xrow = X + (size_t)(tm + r) * ldx;
  const unsigned short* wrow = W + (size_t)(tn + r) * ldw + kofs;

  v8f acc = {};
  for (int k = 0; k < K; k += 32) {
    ABfrag a, b;
    a.h[0] = *(const v8us*)(xrow + k + hi * 8);        // K = k + hi*8 .. +7
    a.h[1] = *(const v8us*)(xrow + k + 16 + hi * 8);   // K = k+16+hi*8 .. +7
    b.h[0] = *(const v8us*)(wrow + k + hi * 8);        // K = k + hi*8 .. +7
    b.h[1] = *(const v8us*)(wrow + k + 16 + hi * 8);   // K = k+16+hi*8 .. +7
    acc = __builtin_amdgcn_wmma_f32_16x16x32_bf16(false, a.bf, false, b.bf,
                                                  (short)0, acc, false, false);
  }
  int n = tn + r;
  float bias = 0.0f;
  if (bias0) bias += bias0[n];
  if (bias1) bias += bias1[n];
#pragma unroll
  for (int j = 0; j < 8; ++j) {
    int m = tm + hi * 8 + j;
    C[(size_t)m * ldc + n] = acc[j] + bias;
  }
}

// ---------------------------------------------------------------------------
// One BiLSTM layer: blockIdx.x = direction (0 fwd, 1 bwd), 512 threads.
// pre[t][j] already holds x@Wih^T + bih + bhh. Whh rows stream from L2.
// h kept in LDS; thread j computes gate j; threads <128 update (c,h).
// ---------------------------------------------------------------------------
__global__ void lstm_layer_kernel(const float* __restrict__ preF, const float* __restrict__ preB,
                                  const float* __restrict__ WhhF, const float* __restrict__ WhhB,
                                  const float* __restrict__ h0, const float* __restrict__ c0,
                                  int h0base, unsigned short* __restrict__ hcat) {
  int dir = blockIdx.x;
  const float* pre = dir ? preB : preF;
  const float* Whh = dir ? WhhB : WhhF;
  __shared__ __align__(16) float hsh[H];
  __shared__ float csh[H];
  __shared__ float gsh[G4];
  int j = threadIdx.x;
  if (j < H) {
    hsh[j] = h0[(size_t)(h0base + dir) * H + j];
    csh[j] = c0[(size_t)(h0base + dir) * H + j];
  }
  __syncthreads();
  const float4* wr4 = (const float4*)(Whh + (size_t)j * H);
  for (int s = 0; s < SEQ; ++s) {
    int t = dir ? (SEQ - 1 - s) : s;
    float acc = pre[(size_t)t * G4 + j];
    const float4* h4 = (const float4*)hsh;
#pragma unroll 8
    for (int kk = 0; kk < H / 4; ++kk) {
      float4 w = wr4[kk], hv = h4[kk];
      acc += w.x * hv.x + w.y * hv.y + w.z * hv.z + w.w * hv.w;
    }
    gsh[j] = acc;
    __syncthreads();
    if (j < H) {
      float ig = sigf(gsh[j]);
      float fg = sigf(gsh[H + j]);
      float gg = tanhf(gsh[2 * H + j]);
      float og = sigf(gsh[3 * H + j]);
      float c  = fg * csh[j] + ig * gg;
      float h  = og * tanhf(c);
      csh[j] = c;
      hsh[j] = h;
      hcat[(size_t)t * DD + dir * H + j] = f2bf(h);
    }
    __syncthreads();
  }
}

// ---------------------------------------------------------------------------
// scores[n,m] = sum_h tanh(A[n,h]+B[m,h]+b1[h])*W2[h] + b2 ; zero diagonal.
// 16x16 (n,m) tile per block; A/B rows staged through padded LDS.
// ---------------------------------------------------------------------------
__global__ void scores_kernel(const float* __restrict__ A, const float* __restrict__ B,
                              const float* __restrict__ b1, const float* __restrict__ W2,
                              const float* __restrict__ b2, float* __restrict__ S) {
  __shared__ float Ash[16][65];
  __shared__ float Bsh[16][65];
  __shared__ float w2s[64];
  __shared__ float b1s[64];
  int tid = threadIdx.x;     // 0..255
  int tx = tid & 15;         // m within tile
  int ty = tid >> 4;         // n within tile
  int n0 = blockIdx.y * 16;
  int m0 = blockIdx.x * 16;
  float s = 0.0f;
  for (int hc = 0; hc < HID; hc += 64) {
    for (int i = tid; i < 16 * 64; i += 256) {
      int rr = i >> 6, cc = i & 63;
      Ash[rr][cc] = A[(size_t)(n0 + rr) * HID + hc + cc];
      Bsh[rr][cc] = B[(size_t)(m0 + rr) * HID + hc + cc];
    }
    if (tid < 64) { w2s[tid] = W2[hc + tid]; b1s[tid] = b1[hc + tid]; }
    __syncthreads();
#pragma unroll 4
    for (int hh = 0; hh < 64; ++hh)
      s += tanhf(Ash[ty][hh] + Bsh[tx][hh] + b1s[hh]) * w2s[hh];
    __syncthreads();
  }
  s += b2[0];
  int n = n0 + ty, m = m0 + tx;
  if (n == m) s = 0.0f;
  S[(size_t)n * SEQ + m] = s;
}

// ---------------------------------------------------------------------------
// Column sums over n for each m.
// ---------------------------------------------------------------------------
__global__ void colsum_kernel(const float* __restrict__ S, float* __restrict__ cs) {
  __shared__ float red[256];
  int m = blockIdx.x, tid = threadIdx.x;
  float p = 0.0f;
  for (int n = tid; n < SEQ; n += 256) p += S[(size_t)n * SEQ + m];
  red[tid] = p;
  __syncthreads();
  for (int o = 128; o > 0; o >>= 1) {
    if (tid < o) red[tid] += red[tid + o];
    __syncthreads();
  }
  if (tid == 0) cs[m] = red[0];
}

// ---------------------------------------------------------------------------
// out[n,:] = softmax_m( S[n,m] / colsum[m] )
// ---------------------------------------------------------------------------
__global__ void softmax_kernel(const float* __restrict__ S, const float* __restrict__ cs,
                               float* __restrict__ out) {
  __shared__ float red[256];
  int n = blockIdx.x, tid = threadIdx.x;
  float v0 = S[(size_t)n * SEQ + tid]       / cs[tid];
  float v1 = S[(size_t)n * SEQ + tid + 256] / cs[tid + 256];
  float mx = fmaxf(v0, v1);
  red[tid] = mx;
  __syncthreads();
  for (int o = 128; o > 0; o >>= 1) {
    if (tid < o) red[tid] = fmaxf(red[tid], red[tid + o]);
    __syncthreads();
  }
  mx = red[0];
  __syncthreads();
  float e0 = __expf(v0 - mx), e1 = __expf(v1 - mx);
  red[tid] = e0 + e1;
  __syncthreads();
  for (int o = 128; o > 0; o >>= 1) {
    if (tid < o) red[tid] += red[tid + o];
    __syncthreads();
  }
  float inv = 1.0f / red[0];
  out[(size_t)n * SEQ + tid]       = e0 * inv;
  out[(size_t)n * SEQ + tid + 256] = e1 * inv;
}

// ---------------------------------------------------------------------------
extern "C" void kernel_launch(void* const* d_in, const int* in_sizes, int n_in,
                              void* d_out, int out_size, void* d_ws, size_t ws_size,
                              hipStream_t stream) {
  (void)in_sizes; (void)n_in; (void)out_size; (void)ws_size;
  const int*   word_ids = (const int*)d_in[0];
  const int*   tag_ids  = (const int*)d_in[1];
  const float* wtab     = (const float*)d_in[2];
  const float* ttab     = (const float*)d_in[3];
  const float* h0       = (const float*)d_in[4];
  const float* c0       = (const float*)d_in[5];
  const float* Wih_l0f  = (const float*)d_in[6];
  const float* Whh_l0f  = (const float*)d_in[7];
  const float* bih_l0f  = (const float*)d_in[8];
  const float* bhh_l0f  = (const float*)d_in[9];
  const float* Wih_l0b  = (const float*)d_in[10];
  const float* Whh_l0b  = (const float*)d_in[11];
  const float* bih_l0b  = (const float*)d_in[12];
  const float* bhh_l0b  = (const float*)d_in[13];
  const float* Wih_l1f  = (const float*)d_in[14];
  const float* Whh_l1f  = (const float*)d_in[15];
  const float* bih_l1f  = (const float*)d_in[16];
  const float* bhh_l1f  = (const float*)d_in[17];
  const float* Wih_l1b  = (const float*)d_in[18];
  const float* Whh_l1b  = (const float*)d_in[19];
  const float* bih_l1b  = (const float*)d_in[20];
  const float* bhh_l1b  = (const float*)d_in[21];
  const float* W1       = (const float*)d_in[22];
  const float* b1       = (const float*)d_in[23];
  const float* W2       = (const float*)d_in[24];
  const float* b2       = (const float*)d_in[25];
  float* out = (float*)d_out;

  char* ws = (char*)d_ws;
  size_t off = 0;
  auto alloc = [&](size_t bytes) {
    char* p = ws + off;
    off += (bytes + 255) & ~(size_t)255;
    return (void*)p;
  };
  unsigned short* xbf   = (unsigned short*)alloc((size_t)SEQ * H * 2);
  unsigned short* w0f   = (unsigned short*)alloc((size_t)G4 * H * 2);
  unsigned short* w0b   = (unsigned short*)alloc((size_t)G4 * H * 2);
  unsigned short* w1f   = (unsigned short*)alloc((size_t)G4 * DD * 2);
  unsigned short* w1b   = (unsigned short*)alloc((size_t)G4 * DD * 2);
  unsigned short* W1bf  = (unsigned short*)alloc((size_t)HID * HID * 2);
  unsigned short* h0cat = (unsigned short*)alloc((size_t)SEQ * DD * 2);
  unsigned short* h1cat = (unsigned short*)alloc((size_t)SEQ * DD * 2);
  float* P0 = (float*)alloc((size_t)SEQ * G4 * 4);   // pre0f -> Amat
  float* P1 = (float*)alloc((size_t)SEQ * G4 * 4);   // pre0b -> Bmat
  float* P2 = (float*)alloc((size_t)SEQ * G4 * 4);   // pre1f -> scores
  float* P3 = (float*)alloc((size_t)SEQ * G4 * 4);   // pre1b
  float* colsum = (float*)alloc((size_t)SEQ * 4);

  // 1. embeddings + weight conversion to bf16
  embed_kernel<<<SEQ, H, 0, stream>>>(word_ids, tag_ids, wtab, ttab, xbf);
  cvt_f32_bf16<<<(G4 * H + 255) / 256, 256, 0, stream>>>(Wih_l0f, w0f, G4 * H);
  cvt_f32_bf16<<<(G4 * H + 255) / 256, 256, 0, stream>>>(Wih_l0b, w0b, G4 * H);
  cvt_f32_bf16<<<(G4 * DD + 255) / 256, 256, 0, stream>>>(Wih_l1f, w1f, G4 * DD);
  cvt_f32_bf16<<<(G4 * DD + 255) / 256, 256, 0, stream>>>(Wih_l1b, w1b, G4 * DD);
  cvt_f32_bf16<<<(HID * HID + 255) / 256, 256, 0, stream>>>(W1, W1bf, HID * HID);

  const int tilesN = G4 / 16;                 // 32
  const int tiles  = (SEQ / 16) * tilesN;     // 1024 tiles, 4 waves/block
  const int gblocks = tiles / 4;              // 256

  // 2. layer-0 input projections (WMMA) + recurrence
  gemm_bf16_wmma<<<gblocks, 128, 0, stream>>>(xbf, H, w0f, H, 0, bih_l0f, bhh_l0f, P0, G4, H, tilesN);
  gemm_bf16_wmma<<<gblocks, 128, 0, stream>>>(xbf, H, w0b, H, 0, bih_l0b, bhh_l0b, P1, G4, H, tilesN);
  lstm_layer_kernel<<<2, G4, 0, stream>>>(P0, P1, Whh_l0f, Whh_l0b, h0, c0, 0, h0cat);

  // 3. layer-1 input projections (WMMA) + recurrence
  gemm_bf16_wmma<<<gblocks, 128, 0, stream>>>(h0cat, DD, w1f, DD, 0, bih_l1f, bhh_l1f, P2, G4, DD, tilesN);
  gemm_bf16_wmma<<<gblocks, 128, 0, stream>>>(h0cat, DD, w1b, DD, 0, bih_l1b, bhh_l1b, P3, G4, DD, tilesN);
  lstm_layer_kernel<<<2, G4, 0, stream>>>(P2, P3, Whh_l1f, Whh_l1b, h0, c0, 2, h1cat);

  // 4. A = h1 @ W1[:, :256]^T ; B = h1 @ W1[:, 256:]^T (WMMA, column offset)
  gemm_bf16_wmma<<<gblocks, 128, 0, stream>>>(h1cat, DD, W1bf, HID, 0,  nullptr, nullptr, P0, HID, DD, tilesN);
  gemm_bf16_wmma<<<gblocks, 128, 0, stream>>>(h1cat, DD, W1bf, HID, DD, nullptr, nullptr, P1, HID, DD, tilesN);

  // 5. pairwise tanh scorer, normalize, softmax
  scores_kernel<<<dim3(32, 32), 256, 0, stream>>>(P0, P1, b1, W2, b2, P2);
  colsum_kernel<<<SEQ, 256, 0, stream>>>(P2, colsum);
  softmax_kernel<<<SEQ, 256, 0, stream>>>(P2, colsum, out);
}
